// DesExpertMoE_49392123904142
// MI455X (gfx1250) — matrix operand order repaired and checked by reading, not compiled, not run on hardware
//
#include <hip/hip_runtime.h>
#include <hip/hip_bf16.h>

typedef _Float16 f16;
typedef __attribute__((ext_vector_type(16))) _Float16 v16h;
typedef __attribute__((ext_vector_type(8)))  float    v8f;

#define BTOK 65536
#define HDIM 768

union FragU { unsigned u[8]; v16h h; };

// ---------------------------------------------------------------------------
// Fused GEMM (+bias, optional LayerNorm+ReLU) using v_wmma_f32_16x16x32_f16.
// Block: WAVES waves (wave32). M-tile = 32 rows per block, each wave owns
// N/WAVES columns. A is [B,KDIM] f16 row-major. Wt is the PRE-TRANSPOSED
// weight, f16 [N][KDIM] row-major. Output f16 [B,N].
// LDS staging: double-buffered, b128 vectorized, staged in pairs of chunks
// (2 loads in flight per group) to bound register pressure (no scratch spills).
// ---------------------------------------------------------------------------
template<int KDIM, int N, int WAVES, bool DO_LN>
__global__ __launch_bounds__(WAVES * 32, 1)
void gemm_ln_k(const f16* __restrict__ A, const f16* __restrict__ Wt,
               const float* __restrict__ bias, const float* __restrict__ gamma,
               const float* __restrict__ beta, f16* __restrict__ out)
{
    constexpr int BLOCK = WAVES * 32;
    constexpr int NW    = N / WAVES;   // columns per wave
    constexpr int NT    = NW / 16;     // 16-wide n-tiles per wave
    constexpr int KS    = 40;          // 32 k + 8 pad (f16); row = 80B (16B mult)

    constexpr unsigned SZ_A  = 32u * KS * 2u;
    constexpr unsigned SZ_B  = (unsigned)N * KS * 2u;
    constexpr unsigned SZ_AB = SZ_A + SZ_B;          // one stage buffer
    constexpr unsigned P1    = 2u * SZ_AB;           // double buffered
    constexpr unsigned SZ_C  = 16u * N * 4u;
    constexpr unsigned P2    = SZ_C + 16u * 16u * 2u * 4u + 16u * 2u * 4u;
    constexpr unsigned SMEM  = P1 > P2 ? P1 : P2;

    constexpr int BVEC = N * 4;        // b128 chunks in the B tile
    constexpr int BIT  = BVEC / BLOCK; // per-thread B staging chunks (even)

    __shared__ alignas(16) char raw[SMEM];
    float* sC = (float*)raw;            // [16][N]      LN spill (after K loop)
    float* sP = (float*)(raw + SZ_C);   // [16][TPR][2] partial sums
    float* sS = sP + 16 * 16 * 2;       // [16][2]      mean / rstd

    const int tid  = threadIdx.x;
    const int wave = tid >> 5;
    const int lane = tid & 31;
    const int hi   = lane >> 4;         // which K half-group this lane carries
    const int l16  = lane & 15;
    const long b0  = (long)blockIdx.x * 32;

    v8f acc[2][NT];
    const v8f vzero = {0.f, 0.f, 0.f, 0.f, 0.f, 0.f, 0.f, 0.f};
#pragma unroll
    for (int mt = 0; mt < 2; ++mt)
#pragma unroll
        for (int nt = 0; nt < NT; ++nt) acc[mt][nt] = vzero;

    // stage tile for k-offset k0 into buffer `buf`
    auto stage = [&](int k0, int buf) {
        f16* sA  = (f16*)(raw + buf * SZ_AB);
        f16* sBt = (f16*)(raw + buf * SZ_AB + SZ_A);
        // A tile: 128 b128 chunks, one per thread (BLOCK >= 128 always)
        if (tid < 128) {
            const int ar = tid >> 2, ac = tid & 3;
            uint4 av = *(const uint4*)(A + (b0 + ar) * KDIM + (k0 + ac * 8));
            *(uint4*)(sA + ar * KS + ac * 8) = av;
        }
        // B tile: pairs of b128 chunks -> 2 loads in flight, low reg pressure
#pragma unroll
        for (int g = 0; g < BIT; g += 2) {
            const int i0 = tid + (g + 0) * BLOCK;
            const int i1 = tid + (g + 1) * BLOCK;
            const int n0 = i0 >> 2, c0 = i0 & 3;
            const int n1 = i1 >> 2, c1 = i1 & 3;
            uint4 v0 = *(const uint4*)(Wt + (size_t)n0 * KDIM + (k0 + c0 * 8));
            uint4 v1 = *(const uint4*)(Wt + (size_t)n1 * KDIM + (k0 + c1 * 8));
            *(uint4*)(sBt + n0 * KS + c0 * 8) = v0;
            *(uint4*)(sBt + n1 * KS + c1 * 8) = v1;
        }
    };

    stage(0, 0);
    __syncthreads();

    for (int k0 = 0; k0 < KDIM; k0 += 32) {
        const int cur = (k0 >> 5) & 1;
        if (k0 + 32 < KDIM) {
            stage(k0 + 32, cur ^ 1);            // overlap with compute below
            __builtin_prefetch((const void*)(A + (b0) * KDIM + k0 + 64), 0, 1);
        }
        const f16* sA  = (const f16*)(raw + cur * SZ_AB);
        const f16* sBt = (const f16*)(raw + cur * SZ_AB + SZ_A);

        // A fragments (ISA 16-bit A 16x32 layout):
        // lane(0..15): row=l16, K = {2j | j<4} and {8+2j | j>=4}; lanes 16..31: +8
        FragU afr[2];
#pragma unroll
        for (int mt = 0; mt < 2; ++mt) {
            const f16* arow = sA + (mt * 16 + l16) * KS + hi * 8;
#pragma unroll
            for (int j = 0; j < 8; ++j) {
                int k = (j < 4) ? (2 * j) : (8 + 2 * j);
                afr[mt].u[j] = *(const unsigned*)(arow + k);
            }
        }
        // B fragments: lane carries column n=l16, K half selected by hi
#pragma unroll
        for (int nt = 0; nt < NT; ++nt) {
            FragU bfr;
            const int n = wave * NW + nt * 16 + l16;
            const f16* brow = sBt + n * KS + hi * 8;
#pragma unroll
            for (int j = 0; j < 8; ++j)
                bfr.u[j] = *(const unsigned*)(brow + ((j < 4) ? (2 * j) : (8 + 2 * j)));
#pragma unroll
            for (int mt = 0; mt < 2; ++mt)
                acc[mt][nt] = __builtin_amdgcn_wmma_f32_16x16x32_f16(
                    false, afr[mt].h, false, bfr.h, (short)0, acc[mt][nt],
                    false, false);
        }
        __syncthreads();   // next iteration's buffer is now fully staged
    }

    // ---------------- epilogue ----------------
    if (!DO_LN) {
        // bias only (expert layer 4)
#pragma unroll
        for (int mt = 0; mt < 2; ++mt)
#pragma unroll
            for (int nt = 0; nt < NT; ++nt) {
                const int n = wave * NW + nt * 16 + l16;
#pragma unroll
                for (int j = 0; j < 8; ++j) {
                    int r = mt * 16 + hi * 8 + j;
                    out[(b0 + r) * N + n] = (f16)(acc[mt][nt][j] + bias[n]);
                }
            }
        return;
    }

    constexpr int TPR   = BLOCK / 16;  // threads cooperating per row
    constexpr int CHUNK = N / TPR;
    for (int mt = 0; mt < 2; ++mt) {
        // spill 16 rows of f32 accum (+bias) to LDS
#pragma unroll
        for (int nt = 0; nt < NT; ++nt) {
            const int n = wave * NW + nt * 16 + l16;
#pragma unroll
            for (int j = 0; j < 8; ++j)
                sC[(hi * 8 + j) * N + n] = acc[mt][nt][j] + bias[n];
        }
        __syncthreads();
        {
            int r = tid / TPR, sub = tid % TPR;
            float s = 0.f, sq = 0.f;
            for (int c = sub * CHUNK; c < (sub + 1) * CHUNK; ++c) {
                float v = sC[r * N + c];
                s += v; sq += v * v;
            }
            sP[(r * TPR + sub) * 2 + 0] = s;
            sP[(r * TPR + sub) * 2 + 1] = sq;
        }
        __syncthreads();
        if (tid < 16) {
            float s = 0.f, sq = 0.f;
            for (int i = 0; i < TPR; ++i) {
                s  += sP[(tid * TPR + i) * 2 + 0];
                sq += sP[(tid * TPR + i) * 2 + 1];
            }
            float mean = s / N;
            float var  = sq / N - mean * mean;   // population variance (ddof=0)
            sS[tid * 2 + 0] = mean;
            sS[tid * 2 + 1] = rsqrtf(var + 1e-5f);
        }
        __syncthreads();
        {
            int r = tid / TPR, sub = tid % TPR;
            float mean = sS[r * 2 + 0], rstd = sS[r * 2 + 1];
            for (int c = sub * CHUNK; c < (sub + 1) * CHUNK; ++c) {
                float v = (sC[r * N + c] - mean) * rstd * gamma[c] + beta[c];
                v = v > 0.f ? v : 0.f;
                out[(b0 + mt * 16 + r) * N + c] = (f16)v;
            }
        }
        __syncthreads();
    }
}

// ---------------------------------------------------------------------------
__global__ void f32tof16(const float* __restrict__ s, f16* __restrict__ d, long n)
{
    long i = (long)blockIdx.x * blockDim.x + threadIdx.x;
    long stride = (long)gridDim.x * blockDim.x;
    for (; i < n; i += stride) d[i] = (f16)s[i];
}

// convert + transpose: src f32 [E][K][N] -> dst f16 [E][N][K]
__global__ void cvt_tr_k(const float* __restrict__ s, f16* __restrict__ d,
                         int K, int N, long total)
{
    long i = (long)blockIdx.x * blockDim.x + threadIdx.x;
    long stride = (long)gridDim.x * blockDim.x;
    long KN = (long)K * N;
    for (; i < total; i += stride) {
        long e = i / KN, r = i % KN;
        int k = (int)(r / N), n = (int)(r % N);
        d[e * KN + (long)n * K + k] = (f16)s[i];
    }
}

// gating head: logits = g2 @ gw3 + gb3 ; softmax ; top-2 ; renormalize
__global__ __launch_bounds__(256)
void gate_finalize_k(const f16* __restrict__ g2, const float* __restrict__ gw3,
                     const float* __restrict__ gb3, float* __restrict__ topw,
                     int* __restrict__ topi)
{
    __shared__ float w[128 * 4];
    for (int i = threadIdx.x; i < 512; i += 256) w[i] = gw3[i];
    __syncthreads();
    long b = (long)blockIdx.x * 256 + threadIdx.x;
    float l[4] = { gb3[0], gb3[1], gb3[2], gb3[3] };
    const f16* h = g2 + b * 128;
    for (int k = 0; k < 128; ++k) {
        float hv = (float)h[k];
#pragma unroll
        for (int e = 0; e < 4; ++e) l[e] += hv * w[k * 4 + e];
    }
    int i0 = 0;
#pragma unroll
    for (int e = 1; e < 4; ++e) if (l[e] > l[i0]) i0 = e;   // ties -> lower idx
    int i1 = -1;
#pragma unroll
    for (int e = 0; e < 4; ++e) {
        if (e == i0) continue;
        if (i1 < 0 || l[e] > l[i1]) i1 = e;
    }
    // full softmax then renormalize over top-2 == softmax over the two logits
    float m  = fmaxf(l[i0], l[i1]);
    float e0 = __expf(l[i0] - m), e1 = __expf(l[i1] - m);
    float w0 = e0 / (e0 + e1);
    topw[b * 2 + 0] = w0;
    topw[b * 2 + 1] = 1.f - w0;
    topi[b * 2 + 0] = i0;
    topi[b * 2 + 1] = i1;
}

// expert_repr[b,d] = sum_k topw[b,k] * eout[topi[b,k], b, d]
__global__ __launch_bounds__(256)
void combine_k(const f16* __restrict__ eout, const float* __restrict__ topw,
               const int* __restrict__ topi, float* __restrict__ out)
{
    long idx = (long)blockIdx.x * 256 + threadIdx.x;  // [0, B*64)
    long b = idx >> 6;
    int  d = (int)(idx & 63);
    int i0 = topi[b * 2 + 0], i1 = topi[b * 2 + 1];
    float w0 = topw[b * 2 + 0], w1 = topw[b * 2 + 1];
    float v = w0 * (float)eout[((long)i0 * BTOK + b) * 64 + d]
            + w1 * (float)eout[((long)i1 * BTOK + b) * 64 + d];
    out[idx] = v;
}

// bot classifier: 64 -> 64 (LN,ReLU) -> 32 (LN,ReLU) -> 1 (sigmoid)
__global__ __launch_bounds__(256)
void classifier_k(const float* __restrict__ xr,
                  const float* cw1, const float* cb1, const float* cg1, const float* cbe1,
                  const float* cw2, const float* cb2, const float* cg2, const float* cbe2,
                  const float* cw3, const float* cb3,
                  float* __restrict__ prob)
{
    __shared__ float sw1[64 * 64], sw2[64 * 32], sw3[32];
    __shared__ float sb1[64], sg1[64], sbe1[64], sb2[32], sg2[32], sbe2[32];
    for (int i = threadIdx.x; i < 64 * 64; i += 256) sw1[i] = cw1[i];
    for (int i = threadIdx.x; i < 64 * 32; i += 256) sw2[i] = cw2[i];
    if (threadIdx.x < 64) {
        sb1[threadIdx.x]  = cb1[threadIdx.x];
        sg1[threadIdx.x]  = cg1[threadIdx.x];
        sbe1[threadIdx.x] = cbe1[threadIdx.x];
    }
    if (threadIdx.x < 32) {
        sw3[threadIdx.x]  = cw3[threadIdx.x];
        sb2[threadIdx.x]  = cb2[threadIdx.x];
        sg2[threadIdx.x]  = cg2[threadIdx.x];
        sbe2[threadIdx.x] = cbe2[threadIdx.x];
    }
    __syncthreads();
    long b = (long)blockIdx.x * 256 + threadIdx.x;

    float x[64];
    for (int d = 0; d < 64; ++d) x[d] = xr[b * 64 + d];

    float h[64];
    for (int o = 0; o < 64; ++o) {
        float s = sb1[o];
        for (int d = 0; d < 64; ++d) s += x[d] * sw1[d * 64 + o];
        h[o] = s;
    }
    float s1 = 0.f, s2 = 0.f;
    for (int o = 0; o < 64; ++o) { s1 += h[o]; s2 += h[o] * h[o]; }
    float m = s1 / 64.f, r = rsqrtf(s2 / 64.f - m * m + 1e-5f);
    for (int o = 0; o < 64; ++o) {
        float t = (h[o] - m) * r * sg1[o] + sbe1[o];
        h[o] = t > 0.f ? t : 0.f;
    }
    float h2[32];
    for (int o = 0; o < 32; ++o) {
        float s = sb2[o];
        for (int d = 0; d < 64; ++d) s += h[d] * sw2[d * 32 + o];
        h2[o] = s;
    }
    s1 = 0.f; s2 = 0.f;
    for (int o = 0; o < 32; ++o) { s1 += h2[o]; s2 += h2[o] * h2[o]; }
    m = s1 / 32.f; r = rsqrtf(s2 / 32.f - m * m + 1e-5f);
    float s = cb3[0];
    for (int o = 0; o < 32; ++o) {
        float t = (h2[o] - m) * r * sg2[o] + sbe2[o];
        t = t > 0.f ? t : 0.f;
        s += t * sw3[o];
    }
    prob[b] = 1.f / (1.f + __expf(-s));
}

// ---------------------------------------------------------------------------
extern "C" void kernel_launch(void* const* d_in, const int* in_sizes, int n_in,
                              void* d_out, int out_size, void* d_ws, size_t ws_size,
                              hipStream_t stream)
{
    const float* x    = (const float*)d_in[0];
    const float* gw1  = (const float*)d_in[1];
    const float* gb1  = (const float*)d_in[2];
    const float* gg1  = (const float*)d_in[3];
    const float* gbe1 = (const float*)d_in[4];
    const float* gw2  = (const float*)d_in[5];
    const float* gb2  = (const float*)d_in[6];
    const float* gg2  = (const float*)d_in[7];
    const float* gbe2 = (const float*)d_in[8];
    const float* gw3  = (const float*)d_in[9];
    const float* gb3  = (const float*)d_in[10];
    const float* ew1  = (const float*)d_in[11];
    const float* eb1  = (const float*)d_in[12];
    const float* eg1  = (const float*)d_in[13];
    const float* ebe1 = (const float*)d_in[14];
    const float* ew2  = (const float*)d_in[15];
    const float* eb2  = (const float*)d_in[16];
    const float* eg2  = (const float*)d_in[17];
    const float* ebe2 = (const float*)d_in[18];
    const float* ew3  = (const float*)d_in[19];
    const float* eb3  = (const float*)d_in[20];
    const float* eg3  = (const float*)d_in[21];
    const float* ebe3 = (const float*)d_in[22];
    const float* ew4  = (const float*)d_in[23];
    const float* eb4  = (const float*)d_in[24];
    const float* cw1  = (const float*)d_in[25];
    const float* cb1  = (const float*)d_in[26];
    const float* cg1  = (const float*)d_in[27];
    const float* cbe1 = (const float*)d_in[28];
    const float* cw2  = (const float*)d_in[29];
    const float* cb2  = (const float*)d_in[30];
    const float* cg2  = (const float*)d_in[31];
    const float* cbe2 = (const float*)d_in[32];
    const float* cw3  = (const float*)d_in[33];
    const float* cb3  = (const float*)d_in[34];

    char* ws = (char*)d_ws;
    size_t off = 0;
    auto take = [&](size_t bytes) -> char* {
        char* p = ws + off;
        off += (bytes + 255) & ~(size_t)255;
        return p;
    };
    f16*   x16  = (f16*)take((size_t)BTOK * HDIM * 2);
    f16*   wg1h = (f16*)take((size_t)768 * 256 * 2);   // transposed [256][768]
    f16*   wg2h = (f16*)take((size_t)256 * 128 * 2);   // transposed [128][256]
    f16*   we1h = (f16*)take((size_t)4 * 768 * 512 * 2); // [E][512][768]
    f16*   we2h = (f16*)take((size_t)4 * 512 * 256 * 2); // [E][256][512]
    f16*   we3h = (f16*)take((size_t)4 * 256 * 128 * 2); // [E][128][256]
    f16*   we4h = (f16*)take((size_t)4 * 128 * 64 * 2);  // [E][64][128]
    f16*   g1b  = (f16*)take((size_t)BTOK * 256 * 2);
    f16*   g2b  = (f16*)take((size_t)BTOK * 128 * 2);
    float* topw = (float*)take((size_t)BTOK * 2 * 4);
    int*   topi = (int*)take((size_t)BTOK * 2 * 4);
    f16*   h1   = (f16*)take((size_t)BTOK * 512 * 2);
    f16*   h2   = (f16*)take((size_t)BTOK * 256 * 2);
    f16*   h3   = (f16*)take((size_t)BTOK * 128 * 2);
    f16*   eoutb= (f16*)take((size_t)4 * BTOK * 64 * 2);

    auto cvt = [&](const float* s, f16* d, long n) {
        long blocks = (n + 255) / 256;
        if (blocks > 32768) blocks = 32768;
        f32tof16<<<dim3((unsigned)blocks), dim3(256), 0, stream>>>(s, d, n);
    };
    auto cvtT = [&](const float* s, f16* d, int K, int N, long total) {
        long blocks = (total + 255) / 256;
        if (blocks > 32768) blocks = 32768;
        cvt_tr_k<<<dim3((unsigned)blocks), dim3(256), 0, stream>>>(s, d, K, N, total);
    };
    cvt (x,   x16, (long)BTOK * HDIM);
    cvtT(gw1, wg1h, 768, 256, 768L * 256);
    cvtT(gw2, wg2h, 256, 128, 256L * 128);
    cvtT(ew1, we1h, 768, 512, 4L * 768 * 512);
    cvtT(ew2, we2h, 512, 256, 4L * 512 * 256);
    cvtT(ew3, we3h, 256, 128, 4L * 256 * 128);
    cvtT(ew4, we4h, 128,  64, 4L * 128 * 64);

    dim3 gemmGrid(BTOK / 32);
    // gating trunk
    gemm_ln_k<768, 256, 8, true><<<gemmGrid, 256, 0, stream>>>(x16, wg1h, gb1, gg1, gbe1, g1b);
    gemm_ln_k<256, 128, 8, true><<<gemmGrid, 256, 0, stream>>>(g1b, wg2h, gb2, gg2, gbe2, g2b);
    gate_finalize_k<<<dim3(BTOK / 256), dim3(256), 0, stream>>>(g2b, gw3, gb3, topw, topi);

    // experts (dense, all E as in reference), one at a time to bound workspace
    for (int e = 0; e < 4; ++e) {
        gemm_ln_k<768, 512, 8, true><<<gemmGrid, 256, 0, stream>>>(
            x16, we1h + (size_t)e * 768 * 512, eb1 + e * 512, eg1 + e * 512, ebe1 + e * 512, h1);
        gemm_ln_k<512, 256, 8, true><<<gemmGrid, 256, 0, stream>>>(
            h1, we2h + (size_t)e * 512 * 256, eb2 + e * 256, eg2 + e * 256, ebe2 + e * 256, h2);
        gemm_ln_k<256, 128, 8, true><<<gemmGrid, 256, 0, stream>>>(
            h2, we3h + (size_t)e * 256 * 128, eb3 + e * 128, eg3 + e * 128, ebe3 + e * 128, h3);
        gemm_ln_k<128, 64, 4, false><<<gemmGrid, 128, 0, stream>>>(
            h3, we4h + (size_t)e * 128 * 64, eb4 + e * 64, nullptr, nullptr,
            eoutb + (size_t)e * BTOK * 64);
    }

    // top-2 weighted combine -> expert_repr (f32, first B*64 of d_out)
    combine_k<<<dim3(BTOK * 64 / 256), dim3(256), 0, stream>>>(eoutb, topw, topi, (float*)d_out);
    // bot classifier -> bot_prob (f32, last B of d_out)
    classifier_k<<<dim3(BTOK / 256), dim3(256), 0, stream>>>(
        (float*)d_out, cw1, cb1, cg1, cbe1, cw2, cb2, cg2, cbe2, cw3, cb3,
        (float*)d_out + (size_t)BTOK * 64);
}
